// Update_88064009437421
// MI455X (gfx1250) — matrix operand order repaired and checked by reading, not compiled, not run on hardware
//
#include <hip/hip_runtime.h>
#include <hip/hip_bf16.h>

// ---------------------------------------------------------------------------
// Sizes (fixed by the reference)
// ---------------------------------------------------------------------------
#define NNODES 32768
#define DD     512
#define DINK   5184
#define SEG    512
#define INVSZ  65536

typedef __attribute__((ext_vector_type(16))) __bf16 v16bf;
typedef __attribute__((ext_vector_type(8)))  __bf16 v8bf;
typedef __attribute__((ext_vector_type(8)))  float  v8f;

static __device__ __forceinline__ float atomicMaxFloat(float* addr, float v) {
    if (v >= 0.f) return __int_as_float(atomicMax((int*)addr, __float_as_int(v)));
    return __uint_as_float(atomicMin((unsigned int*)addr, __float_as_uint(v)));
}

// ---------------------------------------------------------------------------
// WMMA bf16 GEMM:  C[M,N] = epilogue(A[M,K] @ B[K,N] + bias)
//   A,B fp32 row-major, converted to bf16 on the global->LDS path.
//   Block tile 128x128, BK=32, 256 threads = 8 wave32 waves.
//   Wave w computes rows [16w,16w+16) x all 128 cols = 8 WMMA tiles.
//   All 8 B fragments are preloaded so the 8 WMMAs issue back-to-back
//   behind a single s_wait_dscnt (XDL stays busy, ISA §7.5).
// Requires M%128==0, N%128==0, K%32==0 (true for every call here).
// ---------------------------------------------------------------------------
__global__ __launch_bounds__(256)
void k_gemm(const float* __restrict__ A, const float* __restrict__ B,
            const float* __restrict__ bias, const float* residual,
            float* C, int M, int N, int K, int relu)
{
    __shared__ __align__(32) __bf16 sA[128 * 32];
    __shared__ __align__(32) __bf16 sB[32 * 128];

    const int t    = threadIdx.x;
    const int wave = t >> 5;
    const int lane = t & 31;
    const int m0   = blockIdx.y * 128;
    const int n0   = blockIdx.x * 128;

    v8f acc[8];
#pragma unroll
    for (int i = 0; i < 8; ++i) acc[i] = (v8f){};

    const int mrow = wave * 16 + (lane & 15);
    const int hh   = lane >> 4;

    for (int kk = 0; kk < K; kk += 32) {
        if (kk + 32 < K) {  // gfx1250 global_prefetch_b8 hints for next K-chunk
            __builtin_prefetch(A + (size_t)(m0 + (t >> 3)) * K + kk + 32, 0, 3);
            __builtin_prefetch(B + (size_t)(kk + 32 + (t >> 5)) * N + n0, 0, 3);
        }
        // ---- stage A tile [128 x 32] fp32 -> bf16 LDS ----
#pragma unroll
        for (int it = 0; it < 4; ++it) {
            int idx = t + it * 256;               // float4 index, 0..1023
            int r   = idx >> 3;
            int c   = (idx & 7) << 2;
            float4 f = *(const float4*)(A + (size_t)(m0 + r) * K + kk + c);
            __bf16* d = &sA[r * 32 + c];
            d[0] = (__bf16)f.x; d[1] = (__bf16)f.y; d[2] = (__bf16)f.z; d[3] = (__bf16)f.w;
        }
        // ---- stage B tile [32 x 128] fp32 -> bf16 LDS ----
#pragma unroll
        for (int it = 0; it < 4; ++it) {
            int idx = t + it * 256;
            int r   = idx >> 5;
            int c   = (idx & 31) << 2;
            float4 f = *(const float4*)(B + (size_t)(kk + r) * N + n0 + c);
            __bf16* d = &sB[c * 32 + r];
            d[0] = (__bf16)f.x; d[32] = (__bf16)f.y; d[64] = (__bf16)f.z; d[96] = (__bf16)f.w;
        }
        __syncthreads();

        // ---- A fragment (16x32 bf16, ISA 7.12.2 layout) ----
        v8bf alo = *(const v8bf*)&sA[mrow * 32 + hh * 8];
        v8bf ahi = *(const v8bf*)&sA[mrow * 32 + 16 + hh * 8];
        v16bf afrag = __builtin_shufflevector(alo, ahi,
            0, 1, 2, 3, 4, 5, 6, 7, 8, 9, 10, 11, 12, 13, 14, 15);

        // ---- preload all 8 B fragments (lane = N col, 16 bf16 = K in A's order) ----
        v16bf bfrag[8];
#pragma unroll
        for (int nt = 0; nt < 8; ++nt) {
            int nb = (nt * 16 + (lane & 15)) * 32;
            v8bf blo = *(const v8bf*)&sB[nb + hh * 8];
            v8bf bhi = *(const v8bf*)&sB[nb + 16 + hh * 8];
            bfrag[nt] = __builtin_shufflevector(blo, bhi,
                0, 1, 2, 3, 4, 5, 6, 7, 8, 9, 10, 11, 12, 13, 14, 15);
        }

        // ---- 8 back-to-back WMMAs ----
#pragma unroll
        for (int nt = 0; nt < 8; ++nt)
            acc[nt] = __builtin_amdgcn_wmma_f32_16x16x32_bf16(
                false, afrag, false, bfrag[nt], (short)0, acc[nt], false, false);

        __syncthreads();
    }

    // ---- epilogue ----
    const int nlocal = lane & 15;
#pragma unroll
    for (int nt = 0; nt < 8; ++nt) {
        int col = n0 + nt * 16 + nlocal;
        float bb = bias ? bias[col] : 0.f;
#pragma unroll
        for (int j = 0; j < 8; ++j) {
            int row = m0 + wave * 16 + hh * 8 + j;
            float v = acc[nt][j] + bb;
            if (relu) v = fmaxf(v, 0.f);
            if (residual) v += residual[(size_t)row * N + col];
            C[(size_t)row * N + col] = v;
        }
    }
}

// ---------------------------------------------------------------------------
// LayerNorm over D=512: out = LN(x1 (+x2) (+x3)) * g + b, optional relu.
// One 256-thread block per row, 2 elements per thread.
// ---------------------------------------------------------------------------
__global__ __launch_bounds__(256)
void k_layernorm(const float* x1, const float* x2, const float* x3,
                 const float* g, const float* b, float* out, int relu_out)
{
    const int row = blockIdx.x, t = threadIdx.x;
    size_t base = (size_t)row * DD;
    float v0 = x1[base + t], v1 = x1[base + t + 256];
    if (x2) { v0 += x2[base + t]; v1 += x2[base + t + 256]; }
    if (x3) { v0 += x3[base + t]; v1 += x3[base + t + 256]; }

    __shared__ float red[256];
    red[t] = v0 + v1; __syncthreads();
    for (int o = 128; o > 0; o >>= 1) { if (t < o) red[t] += red[t + o]; __syncthreads(); }
    float mu = red[0] * (1.f / DD);
    __syncthreads();
    float d0 = v0 - mu, d1 = v1 - mu;
    red[t] = d0 * d0 + d1 * d1; __syncthreads();
    for (int o = 128; o > 0; o >>= 1) { if (t < o) red[t] += red[t + o]; __syncthreads(); }
    float rstd = rsqrtf(red[0] * (1.f / DD) + 1e-3f);

    float y0 = d0 * rstd * g[t] + b[t];
    float y1 = d1 * rstd * g[t + 256] + b[t + 256];
    if (relu_out) { y0 = fmaxf(y0, 0.f); y1 = fmaxf(y1, 0.f); }
    out[base + t] = y0; out[base + t + 256] = y1;
}

// ---------------------------------------------------------------------------
// Small utility / elementwise kernels (flat id over N*D unless noted)
// ---------------------------------------------------------------------------
__global__ void k_fill_f32(float* p, float v, int n) {
    int i = blockIdx.x * 256 + threadIdx.x; if (i < n) p[i] = v;
}
__global__ void k_fill_i32(int* p, int v, int n) {
    int i = blockIdx.x * 256 + threadIdx.x; if (i < n) p[i] = v;
}
__global__ void k_tmax(const int* tf, int* tmax, int n) {
    int i = blockIdx.x * 256 + threadIdx.x; if (i < n) atomicMax(tmax, tf[i]);
}
__global__ void k_inv(const int* p, const int* tf, const int* tmax, int* inv, int n) {
    int i = blockIdx.x * 256 + threadIdx.x; if (i >= n) return;
    int key = p[i] * (*tmax + 1) + tf[i];
    if (key >= 0 && key < INVSZ) inv[key] = i;
}
__global__ void k_prevnext(const int* p, const int* tf, const int* tmax,
                           const int* inv, int* prv, int* nxt, int n) {
    int i = blockIdx.x * 256 + threadIdx.x; if (i >= n) return;
    int key = p[i] * (*tmax + 1) + tf[i];
    prv[i] = (key >= 1 && key < INVSZ) ? inv[key - 1] : -1;
    nxt[i] = (key + 1 < INVSZ && key >= 0) ? inv[key + 1] : -1;
}
__global__ void k_eid(const int* s, const int* tf, const int* tmax, int* eid, int n) {
    int i = blockIdx.x * 256 + threadIdx.x; if (i < n) eid[i] = s[i] * (*tmax + 1) + tf[i];
}
__global__ void k_gather_mask(const float* H, const int* idx, float* out) {
    size_t id = (size_t)blockIdx.x * 256 + threadIdx.x;
    int r = (int)(id >> 9), d = (int)(id & 511);
    int j = idx[r];
    out[id] = (j >= 1) ? H[(size_t)j * DD + d] : 0.f;   // reference masks idx>=1
}
__global__ void k_segmax(const float* s, const int* gid, float* m) {
    size_t id = (size_t)blockIdx.x * 256 + threadIdx.x;
    int r = (int)(id >> 9), d = (int)(id & 511);
    atomicMaxFloat(&m[gid[r] * DD + d], s[id]);
}
__global__ void k_expdenom(const float* s, const int* gid, const float* m,
                           float* e, float* denom) {
    size_t id = (size_t)blockIdx.x * 256 + threadIdx.x;
    int r = (int)(id >> 9), d = (int)(id & 511);
    float v = __expf(s[id] - m[gid[r] * DD + d]);
    e[id] = v;
    atomicAdd(&denom[gid[r] * DD + d], v);
}
__global__ void k_scattery(const float* tv, const float* e, const float* denom,
                           const int* gid, float* y) {
    size_t id = (size_t)blockIdx.x * 256 + threadIdx.x;
    int r = (int)(id >> 9), d = (int)(id & 511);
    int g = gid[r] * DD + d;
    atomicAdd(&y[g], tv[id] * e[id] / denom[g]);
}
__global__ void k_gatheradd(float* H, const float* z, const int* gid) {
    size_t id = (size_t)blockIdx.x * 256 + threadIdx.x;
    int r = (int)(id >> 9), d = (int)(id & 511);
    H[id] += z[gid[r] * DD + d];
}
__global__ void k_gatecombine(const float* hn, const float* g, const float* res, float* H) {
    size_t id = (size_t)blockIdx.x * 256 + threadIdx.x;
    H[id] = hn[id] + (1.f / (1.f + __expf(-g[id]))) * res[id];
}

// ---------------------------------------------------------------------------
// Heads: delta = relu(h)@Wd+bd ; weights = sigmoid(relu(h)@Ww+bw)
// ---------------------------------------------------------------------------
__global__ __launch_bounds__(256)
void k_head(const float* H, const float* Wd, const float* bd,
            const float* Ww, const float* bw, float* delta, float* wout)
{
    const int row = blockIdx.x, t = threadIdx.x;
    float a0 = 0, a1 = 0, a2 = 0, a3 = 0;
    for (int d = t; d < DD; d += 256) {
        float hv = fmaxf(H[(size_t)row * DD + d], 0.f);
        a0 += hv * Wd[d * 2 + 0]; a1 += hv * Wd[d * 2 + 1];
        a2 += hv * Ww[d * 2 + 0]; a3 += hv * Ww[d * 2 + 1];
    }
    __shared__ float r0[256], r1[256], r2[256], r3[256];
    r0[t] = a0; r1[t] = a1; r2[t] = a2; r3[t] = a3; __syncthreads();
    for (int o = 128; o > 0; o >>= 1) {
        if (t < o) { r0[t] += r0[t+o]; r1[t] += r1[t+o]; r2[t] += r2[t+o]; r3[t] += r3[t+o]; }
        __syncthreads();
    }
    if (t == 0) {
        delta[row * 2 + 0] = r0[0] + bd[0];
        delta[row * 2 + 1] = r1[0] + bd[1];
        wout[row * 2 + 0]  = 1.f / (1.f + __expf(-(r2[0] + bw[0])));
        wout[row * 2 + 1]  = 1.f / (1.f + __expf(-(r3[0] + bw[1])));
    }
}

// ---------------------------------------------------------------------------
// Host-side orchestration
// ---------------------------------------------------------------------------
extern "C" void kernel_launch(void* const* d_in, const int* in_sizes, int n_in,
                              void* d_out, int out_size, void* d_ws, size_t ws_size,
                              hipStream_t stream)
{
    const float* in_h    = (const float*)d_in[0];
    const float* in_corr = (const float*)d_in[2];
    const float* in_ctx  = (const float*)d_in[3];
    const int*   sfi     = (const int*)d_in[4];
    const int*   tfi     = (const int*)d_in[5];
    const int*   pidx    = (const int*)d_in[6];
#define PW(i) ((const float*)d_in[i])

    float* H     = (float*)d_out;                      // [N, D]
    float* delta = H + (size_t)NNODES * DD;            // [N, 2]
    float* wout  = delta + (size_t)NNODES * 2;         // [N, 2]

    // workspace carve-up
    char*  ws  = (char*)d_ws;
    size_t off = 0;
    auto carve = [&](size_t bytes) { void* p = ws + off; off += (bytes + 255) & ~(size_t)255; return p; };
    const size_t BIG = (size_t)NNODES * DD * sizeof(float);
    float* t1   = (float*)carve(BIG);
    float* t2   = (float*)carve(BIG);
    float* t3   = (float*)carve(BIG);
    float* t4   = (float*)carve(BIG);
    float* segm = (float*)carve((size_t)SEG * DD * sizeof(float));
    float* segd = (float*)carve((size_t)SEG * DD * sizeof(float));
    float* segy = (float*)carve((size_t)SEG * DD * sizeof(float));
    float* segz = (float*)carve((size_t)SEG * DD * sizeof(float));
    int*   inv  = (int*)carve(INVSZ * sizeof(int));
    int*   prvI = (int*)carve(NNODES * sizeof(int));
    int*   nxtI = (int*)carve(NNODES * sizeof(int));
    int*   eid  = (int*)carve(NNODES * sizeof(int));
    int*   tmax = (int*)carve(256);

    const float NEG_INF = -__builtin_inff();   // host-side constant (0xff800000)

    const dim3 gBig(DD / 128, NNODES / 128);   // node GEMMs
    const dim3 gSeg(DD / 128, SEG / 128);      // 512x512 GEMM
    const int  EW  = (NNODES * DD) / 256;      // elementwise grid
    const int  EN  = (NNODES + 255) / 256;

    // ---- corr_net: t1=relu(corr@W1+b); t2=t1@W2+b; t1=relu(LN(t2)); c->t2 ----
    k_gemm<<<gBig, 256, 0, stream>>>(in_corr, PW(7), PW(8), nullptr, t1, NNODES, DD, DINK, 1);
    k_gemm<<<gBig, 256, 0, stream>>>(t1, PW(9), PW(10), nullptr, t2, NNODES, DD, DD, 0);
    k_layernorm<<<NNODES, 256, 0, stream>>>(t2, nullptr, nullptr, PW(11), PW(12), t1, 1);
    k_gemm<<<gBig, 256, 0, stream>>>(t1, PW(13), PW(14), nullptr, t2, NNODES, DD, DD, 0);

    // ---- h = LN(h + ctx + c, norm) ----
    k_layernorm<<<NNODES, 256, 0, stream>>>(in_h, in_ctx, t2, PW(15), PW(16), H, 0);

    // ---- neighbours via inverse key map ----
    k_fill_i32<<<(INVSZ + 255) / 256, 256, 0, stream>>>(inv, -1, INVSZ);
    k_fill_i32<<<1, 32, 0, stream>>>(tmax, 0, 1);
    k_tmax<<<EN, 256, 0, stream>>>(tfi, tmax, NNODES);
    k_inv<<<EN, 256, 0, stream>>>(pidx, tfi, tmax, inv, NNODES);
    k_prevnext<<<EN, 256, 0, stream>>>(pidx, tfi, tmax, inv, prvI, nxtI, NNODES);
    k_eid<<<EN, 256, 0, stream>>>(sfi, tfi, tmax, eid, NNODES);

    // ---- h += mlp2(mask*h[prev]); h += mlp2(mask*h[next]) ----
    k_gather_mask<<<EW, 256, 0, stream>>>(H, prvI, t1);
    k_gemm<<<gBig, 256, 0, stream>>>(t1, PW(17), PW(18), nullptr, t2, NNODES, DD, DD, 1);
    k_gemm<<<gBig, 256, 0, stream>>>(t2, PW(19), PW(20), H, H, NNODES, DD, DD, 0);
    k_gather_mask<<<EW, 256, 0, stream>>>(H, nxtI, t1);
    k_gemm<<<gBig, 256, 0, stream>>>(t1, PW(21), PW(22), nullptr, t2, NNODES, DD, DD, 1);
    k_gemm<<<gBig, 256, 0, stream>>>(t2, PW(23), PW(24), H, H, NNODES, DD, DD, 0);

    // ---- soft_agg x2 (pa over patches, ea over edge ids) ----
    const int* gids[2] = { pidx, eid };
    const int  wbase[2] = { 25, 31 };
    for (int a = 0; a < 2; ++a) {
        const int w = wbase[a]; const int* gid = gids[a];
        k_gemm<<<gBig, 256, 0, stream>>>(H, PW(w), PW(w + 1), nullptr, t1, NNODES, DD, DD, 0);
        k_fill_f32<<<(SEG * DD) / 256, 256, 0, stream>>>(segm, NEG_INF, SEG * DD);
        k_fill_f32<<<(SEG * DD) / 256, 256, 0, stream>>>(segd, 0.f, SEG * DD);
        k_fill_f32<<<(SEG * DD) / 256, 256, 0, stream>>>(segy, 0.f, SEG * DD);
        k_segmax<<<EW, 256, 0, stream>>>(t1, gid, segm);
        k_expdenom<<<EW, 256, 0, stream>>>(t1, gid, segm, t2, segd);
        k_gemm<<<gBig, 256, 0, stream>>>(H, PW(w + 2), PW(w + 3), nullptr, t3, NNODES, DD, DD, 0);
        k_scattery<<<EW, 256, 0, stream>>>(t3, t2, segd, gid, segy);
        k_gemm<<<gSeg, 256, 0, stream>>>(segy, PW(w + 4), PW(w + 5), nullptr, segz, SEG, DD, DD, 0);
        k_gatheradd<<<EW, 256, 0, stream>>>(H, segz, gid);
    }

    // ---- two gated residual blocks ----
    const int lnp[2] = { 37, 45 };
    const int grp[2] = { 39, 47 };
    for (int a = 0; a < 2; ++a) {
        k_layernorm<<<NNODES, 256, 0, stream>>>(H, nullptr, nullptr, PW(lnp[a]), PW(lnp[a] + 1), t1, 0);
        k_gemm<<<gBig, 256, 0, stream>>>(t1, PW(grp[a]), PW(grp[a] + 1), nullptr, t2, NNODES, DD, DD, 0); // gate pre
        k_gemm<<<gBig, 256, 0, stream>>>(t1, PW(grp[a] + 2), PW(grp[a] + 3), nullptr, t3, NNODES, DD, DD, 1); // relu(r1)
        k_gemm<<<gBig, 256, 0, stream>>>(t3, PW(grp[a] + 4), PW(grp[a] + 5), nullptr, t4, NNODES, DD, DD, 0); // r2
        k_gatecombine<<<EW, 256, 0, stream>>>(t1, t2, t4, H);
    }

    // ---- heads ----
    k_head<<<NNODES, 256, 0, stream>>>(H, PW(53), PW(54), PW(55), PW(56), delta, wout);
#undef PW
}
